// CoreEncoder_7911329759357
// MI455X (gfx1250) — compile-verified, matrix operand order checked by reading, not run on hardware
//
#include <hip/hip_runtime.h>

// ---------------------------------------------------------------------------
// LPCNet-style encoder for MI455X (gfx1250, wave32, WMMA).
//   X activation buffer: [25600 rows][864 ch] fp16, each stage writes its slice.
//   All GEMMs: v_wmma_f32_16x16x32_f16, wave = 16x16 tile.
//   GRU recurrence: 4 blocks x 16 batches, 12 waves = 12 gate tiles, h in LDS,
//   w_hh tiles held in VGPRs across all 400 serial steps (critical path).
// ---------------------------------------------------------------------------

#define ROWS   25600      // B*T = 64*400
#define TSTEPS 400
#define LDXC   864        // channels in big activation buffer

typedef _Float16 v16h __attribute__((ext_vector_type(16)));
typedef float    v8f  __attribute__((ext_vector_type(8)));
typedef unsigned int u32x4 __attribute__((ext_vector_type(4)));

union HQ { u32x4 q; _Float16 h[8]; };

// Load one operand's worth of a 16x32 (or 32x16) fp16 WMMA tile for this lane.
// p points at element [m][k0 + sub] of a row; layout per CDNA5 ISA 7.12.2:
//   lanes 0-15  (sub=0): K = k0..k0+7 in VGPR0-3, K = k0+16..k0+23 in VGPR4-7
//   lanes 16-31 (sub=8): K = k0+8..k0+15,          K = k0+24..k0+31
static __device__ __forceinline__ v16h ld_tile(const _Float16* p) {
  HQ a, b;
  a.q = *reinterpret_cast<const u32x4*>(p);        // 16B -> b128 load
  b.q = *reinterpret_cast<const u32x4*>(p + 16);   // 16B -> b128 load
  v16h v;
#pragma unroll
  for (int i = 0; i < 8; ++i) { v[i] = a.h[i]; v[i + 8] = b.h[i]; }
  return v;
}

// ------------------------- small prep kernels ------------------------------

__global__ void pad_features(const float* __restrict__ f, _Float16* __restrict__ xf) {
  int i = blockIdx.x * 256 + threadIdx.x;          // over ROWS*64
  if (i >= ROWS * 64) return;
  int row = i >> 6, c = i & 63;
  xf[i] = (_Float16)(c < 40 ? f[row * 40 + c] : 0.0f);  // fold 2 frames, pad K->64
}

__global__ void cvt_f16_pad(const float* __restrict__ src, _Float16* __restrict__ dst,
                            int N, int Kin, int Kout) {
  int i = blockIdx.x * 256 + threadIdx.x;
  if (i >= N * Kout) return;
  int n = i / Kout, k = i % Kout;
  dst[i] = (_Float16)(k < Kin ? src[n * Kin + k] : 0.0f);
}

// conv w: [96][I][2] fp32 -> [96][2I] fp16, tap-major (tap0 cols 0..I-1)
__global__ void repack_conv_w(const float* __restrict__ src, _Float16* __restrict__ dst, int I) {
  int i = blockIdx.x * 256 + threadIdx.x;
  if (i >= 96 * 2 * I) return;
  int o = i / (2 * I), r = i % (2 * I);
  int tap = r / I, c = r % I;
  dst[i] = (_Float16)src[(o * I + c) * 2 + tap];
}

// ------------------------- generic WMMA GEMM -------------------------------
// out[row][n] = act(bias[n] + sum_k A[row][k] * W[n][k])
// one wave = one 16x16 tile; grid = (rowTiles/8, ceil(N/16)); block = 256.
__global__ __launch_bounds__(256)
void wmma_gemm(const _Float16* __restrict__ A, int ldA,
               const _Float16* __restrict__ W, int K,
               const float* __restrict__ bias, int N,
               float* __restrict__ outF, int ldOutF,
               _Float16* __restrict__ outH, int ldOutH, int outColH) {
  const int lane = threadIdx.x & 31;
  const int wave = threadIdx.x >> 5;
  const int row0 = (blockIdx.x * 8 + wave) * 16;
  const int n0   = blockIdx.y * 16;
  const int m    = lane & 15;
  const int sub  = (lane >> 4) * 8;
  const int n    = n0 + m;
  const bool nv  = (n < N);
  const int nsafe = nv ? n : (N - 1);

  const _Float16* aB = A + (size_t)(row0 + m) * ldA + sub;
  const _Float16* wB = W + (size_t)nsafe * K + sub;

  float b0 = nv ? bias[n] : 0.0f;
  v8f acc;
#pragma unroll
  for (int i = 0; i < 8; ++i) acc[i] = b0;

  for (int k0 = 0; k0 < K; k0 += 32) {
    v16h a = ld_tile(aB + k0);
    v16h b = ld_tile(wB + k0);
    acc = __builtin_amdgcn_wmma_f32_16x16x32_f16(false, a, false, b,
                                                 (short)0, acc, false, false);
  }

  const int mhi = (lane >> 4) * 8;   // C/D layout: lanes 16-31 hold M = 8..15
  if (outF) {
    if (nv) {
#pragma unroll
      for (int r = 0; r < 8; ++r)
        outF[(size_t)(row0 + mhi + r) * ldOutF + n] = acc[r];
    }
  } else if (nv) {
#pragma unroll
    for (int r = 0; r < 8; ++r) {
      float v = tanhf(acc[r]);
      v = fminf(1.0f, fmaxf(-1.0f, v));   // _qn clamp (dither omitted)
      outH[(size_t)(row0 + mhi + r) * ldOutH + outColH + n] = (_Float16)v;
    }
  }
}

// ------------------------- causal dilated conv (k=2) as GEMM ---------------
// y[row][o] = tanh(b[o] + sum_i X[row-d][i]*Wc[o][i] + X[row][i]*Wc[o][I+i])
__global__ __launch_bounds__(256)
void conv_gemm(const _Float16* __restrict__ X, int I, int dil,
               const _Float16* __restrict__ Wc, const float* __restrict__ bias,
               _Float16* __restrict__ Xout, int outCol) {
  const int lane = threadIdx.x & 31;
  const int wave = threadIdx.x >> 5;
  const int row0 = (blockIdx.x * 8 + wave) * 16;
  const int n0   = blockIdx.y * 16;      // N = 96, always full tiles
  const int m    = lane & 15;
  const int sub  = (lane >> 4) * 8;
  const int n    = n0 + m;

  const int row_m = row0 + m;
  const int t     = row_m % TSTEPS;       // tiles never cross batch (400 % 16 == 0)
  const bool v0   = (t >= dil);           // tap-0 causal validity (per lane)
  const _Float16* a0 = X + (size_t)(v0 ? (row_m - dil) : row_m) * LDXC + sub;
  const _Float16* a1 = X + (size_t)row_m * LDXC + sub;
  const _Float16* wB = Wc + (size_t)n * (2 * I) + sub;

  float b0 = bias[n];
  v8f acc;
#pragma unroll
  for (int i = 0; i < 8; ++i) acc[i] = b0;

  for (int k0 = 0; k0 < I; k0 += 32) {    // tap 0 (shifted time)
    v16h a = ld_tile(a0 + k0);
#pragma unroll
    for (int i = 0; i < 16; ++i) a[i] = v0 ? a[i] : (_Float16)0;  // zero-pad, no branch
    v16h b = ld_tile(wB + k0);
    acc = __builtin_amdgcn_wmma_f32_16x16x32_f16(false, a, false, b,
                                                 (short)0, acc, false, false);
  }
  for (int k0 = 0; k0 < I; k0 += 32) {    // tap 1 (current time)
    v16h a = ld_tile(a1 + k0);
    v16h b = ld_tile(wB + I + k0);
    acc = __builtin_amdgcn_wmma_f32_16x16x32_f16(false, a, false, b,
                                                 (short)0, acc, false, false);
  }

  const int mhi = (lane >> 4) * 8;
#pragma unroll
  for (int r = 0; r < 8; ++r) {
    float v = tanhf(acc[r]);
    v = fminf(1.0f, fmaxf(-1.0f, v));
    Xout[(size_t)(row0 + mhi + r) * LDXC + outCol + n] = (_Float16)v;
  }
}

// ------------------------- GRU recurrence ----------------------------------
// One block = 16 batches, 12 waves (one per 16-col tile of 192 gate outputs).
// h (fp16) and gate preacts (fp32) live in LDS; 400 serial steps.
// Each wave's two w_hh B-tiles (32 fp16/lane) are loaded ONCE into VGPRs and
// reused for all steps -> no global loads / load-waits on the serial chain.
__global__ __launch_bounds__(384)
void gru_rec(const float* __restrict__ xg,        // [ROWS][192] = b_ih + x@Wih^T
             const _Float16* __restrict__ whh,    // [192][64] fp16
             const float* __restrict__ bhh,       // [192]
             _Float16* __restrict__ X, int outCol) {
  __shared__ alignas(16) _Float16 hbuf[16][64];
  __shared__ float hg[16][192];

  const int tid  = threadIdx.x;
  const int lane = tid & 31;
  const int wave = tid >> 5;
  const int b0   = blockIdx.x * 16;

  for (int e = tid; e < 16 * 64; e += 384) hbuf[e >> 6][e & 63] = (_Float16)0;
  __syncthreads();

  const int m   = lane & 15;
  const int sub = (lane >> 4) * 8;
  const int mhi = (lane >> 4) * 8;
  const int n   = wave * 16 + m;             // gate column 0..191
  const float bn = bhh[n];
  const _Float16* wrow = whh + (size_t)n * 64 + sub;
  const v16h w0 = ld_tile(wrow);             // hoisted: live in VGPRs for all t
  const v16h w1 = ld_tile(wrow + 32);

  for (int t = 0; t < TSTEPS; ++t) {
    v8f acc;
#pragma unroll
    for (int i = 0; i < 8; ++i) acc[i] = bn;

    // hg = b_hh + h @ Whh^T   (K = 64 -> two 16x16x32 WMMAs, A from LDS)
    v16h a0 = ld_tile(&hbuf[m][sub]);
    acc = __builtin_amdgcn_wmma_f32_16x16x32_f16(false, a0, false, w0,
                                                 (short)0, acc, false, false);
    v16h a1 = ld_tile(&hbuf[m][32 + sub]);
    acc = __builtin_amdgcn_wmma_f32_16x16x32_f16(false, a1, false, w1,
                                                 (short)0, acc, false, false);

#pragma unroll
    for (int r = 0; r < 8; ++r) hg[mhi + r][n] = acc[r];
    __syncthreads();

    // gate combine + state update: 16x64 elements over 384 threads
    for (int e = tid; e < 16 * 64; e += 384) {
      int mm = e >> 6, j = e & 63;
      size_t roff = ((size_t)(b0 + mm) * TSTEPS + t) * 192;
      float xr = xg[roff + j];
      float xz = xg[roff + 64 + j];
      float xn = xg[roff + 128 + j];
      float r = 1.0f / (1.0f + __expf(-(xr + hg[mm][j])));
      float z = 1.0f / (1.0f + __expf(-(xz + hg[mm][64 + j])));
      float c = tanhf(xn + r * hg[mm][128 + j]);   // r gates only recurrent part
      float hp = (float)hbuf[mm][j];
      float h  = (1.0f - z) * c + z * hp;
      h = fminf(1.0f, fmaxf(-1.0f, h));
      hbuf[mm][j] = (_Float16)h;
      X[((size_t)(b0 + mm) * TSTEPS + t) * LDXC + outCol + j] = (_Float16)h;
    }
    __syncthreads();
  }
}

// ------------------------- host orchestration ------------------------------

extern "C" void kernel_launch(void* const* d_in, const int* in_sizes, int n_in,
                              void* d_out, int out_size, void* d_ws, size_t ws_size,
                              hipStream_t stream) {
  (void)in_sizes; (void)n_in; (void)out_size; (void)ws_size;

  static const int GRU_IN[5]  = {64, 224, 384, 544, 704};
  static const int CONV_IN[5] = {128, 288, 448, 608, 768};
  static const int DILV[5]    = {1, 2, 2, 2, 2};

  const float* features = (const float*)d_in[0];
  const float* dense1_w = (const float*)d_in[1];
  const float* dense1_b = (const float*)d_in[2];
  const float *wih_f[5], *whh_f[5], *bih[5], *bhh[5], *cw_f[5], *cb[5];
  for (int i = 0; i < 5; ++i) {
    wih_f[i] = (const float*)d_in[3 + 6 * i];
    whh_f[i] = (const float*)d_in[4 + 6 * i];
    bih[i]   = (const float*)d_in[5 + 6 * i];
    bhh[i]   = (const float*)d_in[6 + 6 * i];
    cw_f[i]  = (const float*)d_in[7 + 6 * i];
    cb[i]    = (const float*)d_in[8 + 6 * i];
  }
  const float* z_w  = (const float*)d_in[33];
  const float* z_b  = (const float*)d_in[34];
  const float* s1_w = (const float*)d_in[35];
  const float* s1_b = (const float*)d_in[36];
  const float* s2_w = (const float*)d_in[37];
  const float* s2_b = (const float*)d_in[38];

  // workspace bump allocator
  char* ws = (char*)d_ws;
  size_t off = 0;
  auto alloc = [&](size_t bytes) -> void* {
    void* p = ws + off;
    off = (off + bytes + 255) & ~(size_t)255;
    return p;
  };
  _Float16* X   = (_Float16*)alloc((size_t)ROWS * LDXC * 2);   // big activation buf
  _Float16* xf  = (_Float16*)alloc((size_t)ROWS * 64 * 2);     // folded features
  float*    xg  = (float*)   alloc((size_t)ROWS * 192 * 4);    // gru input proj
  _Float16* s1h = (_Float16*)alloc((size_t)ROWS * 128 * 2);    // s1 hidden
  _Float16* d1w = (_Float16*)alloc(64 * 64 * 2);
  _Float16 *wih16[5], *whh16[5], *cw16[5];
  for (int i = 0; i < 5; ++i) {
    wih16[i] = (_Float16*)alloc((size_t)192 * GRU_IN[i] * 2);
    whh16[i] = (_Float16*)alloc(192 * 64 * 2);
    cw16[i]  = (_Float16*)alloc((size_t)96 * 2 * CONV_IN[i] * 2);
  }
  _Float16* zw16  = (_Float16*)alloc(80 * 864 * 2);
  _Float16* s1w16 = (_Float16*)alloc(128 * 864 * 2);
  _Float16* s2w16 = (_Float16*)alloc(24 * 128 * 2);

  auto cvt = [&](const float* s, _Float16* d, int N, int Kin, int Kout) {
    int tot = N * Kout;
    cvt_f16_pad<<<(tot + 255) / 256, 256, 0, stream>>>(s, d, N, Kin, Kout);
  };

  // weight conversions (deterministic every call)
  pad_features<<<(ROWS * 64 + 255) / 256, 256, 0, stream>>>(features, xf);
  cvt(dense1_w, d1w, 64, 40, 64);
  for (int i = 0; i < 5; ++i) {
    cvt(wih_f[i], wih16[i], 192, GRU_IN[i], GRU_IN[i]);
    cvt(whh_f[i], whh16[i], 192, 64, 64);
    int tot = 96 * 2 * CONV_IN[i];
    repack_conv_w<<<(tot + 255) / 256, 256, 0, stream>>>(cw_f[i], cw16[i], CONV_IN[i]);
  }
  cvt(z_w, zw16, 80, 864, 864);
  cvt(s1_w, s1w16, 128, 864, 864);
  cvt(s2_w, s2w16, 24, 128, 128);

  const dim3 blk(256);
  const int rowBlocks = (ROWS / 16) / 8;   // 200

  // dense1: tanh(xf @ W^T + b) -> X[:, 0:64]
  wmma_gemm<<<dim3(rowBlocks, 4), blk, 0, stream>>>(
      xf, 64, d1w, 64, dense1_b, 64, nullptr, 0, X, LDXC, 0);

  for (int i = 0; i < 5; ++i) {
    // input projection: xg = b_ih + X[:, :GRU_IN] @ Wih^T   (fp32 out)
    wmma_gemm<<<dim3(rowBlocks, 12), blk, 0, stream>>>(
        X, LDXC, wih16[i], GRU_IN[i], bih[i], 192, xg, 192, nullptr, 0, 0);
    // recurrence -> X[:, GRU_IN : GRU_IN+64]
    gru_rec<<<dim3(4), dim3(384), 0, stream>>>(xg, whh16[i], bhh[i], X, GRU_IN[i]);
    // conv -> X[:, CONV_IN : CONV_IN+96]
    conv_gemm<<<dim3(rowBlocks, 6), blk, 0, stream>>>(
        X, CONV_IN[i], DILV[i], cw16[i], cb[i], X, CONV_IN[i]);
  }

  float* zOut = (float*)d_out;                        // [ROWS][80]
  float* sOut = (float*)d_out + (size_t)ROWS * 80;    // [ROWS][24]
  wmma_gemm<<<dim3(rowBlocks, 5), blk, 0, stream>>>(
      X, LDXC, zw16, 864, z_b, 80, zOut, 80, nullptr, 0, 0);
  wmma_gemm<<<dim3(rowBlocks, 8), blk, 0, stream>>>(
      X, LDXC, s1w16, 864, s1_b, 128, nullptr, 0, s1h, 128, 0);
  wmma_gemm<<<dim3(rowBlocks, 2), blk, 0, stream>>>(
      s1h, 128, s2w16, 128, s2_b, 24, sOut, 24, nullptr, 0, 0);
}
